// DomainClassifier_90555090469561
// MI455X (gfx1250) — matrix operand order repaired and checked by reading, not compiled, not run on hardware
//
#include <hip/hip_runtime.h>
#include <math.h>

// ---------------- problem constants (match reference) ----------------
#define N_NODES  100000
#define N_EDGES  1600000
#define E_TOT    (N_EDGES + N_NODES)   // with self loops
#define G_GRAPHS 64
#define IN_DIM   256
#define HID      64
#define HEADS    4
#define F1       (HEADS * HID)         // 256
#define NEG_SLOPE 0.2f

typedef __attribute__((ext_vector_type(16))) __bf16 v16bf;
typedef __attribute__((ext_vector_type(8)))  float  v8f;

// ---------------- helpers ----------------
static __device__ __forceinline__ __bf16 f2bf(float f) {
  unsigned u = __float_as_uint(f);
  unsigned r = (u + 0x7FFFu + ((u >> 16) & 1u)) >> 16;   // RNE
  unsigned short s = (unsigned short)r;
  return __builtin_bit_cast(__bf16, s);
}
// order-preserving float -> uint key (for atomicMax-based segment max)
static __device__ __forceinline__ unsigned fkey(float f) {
  unsigned u = __float_as_uint(f);
  return (u & 0x80000000u) ? ~u : (u | 0x80000000u);
}
static __device__ __forceinline__ float fdec(unsigned k) {
  unsigned u = (k & 0x80000000u) ? (k & 0x7FFFFFFFu) : ~k;
  return __uint_as_float(u);
}

// ---------------- generic fills ----------------
__global__ void fill_f32(float* __restrict__ p, float v, size_t n) {
  size_t t = (size_t)blockIdx.x * blockDim.x + threadIdx.x;
  if (t < n) p[t] = v;
}
__global__ void fill_u32(unsigned* __restrict__ p, unsigned v, size_t n) {
  size_t t = (size_t)blockIdx.x * blockDim.x + threadIdx.x;
  if (t < n) p[t] = v;
}

// ---------------- precision conversion ----------------
__global__ void cvt_f32_bf16(const float* __restrict__ in, __bf16* __restrict__ out, size_t n) {
  size_t t = (size_t)blockIdx.x * blockDim.x + threadIdx.x;
  if (t < n) out[t] = f2bf(in[t]);
}
// W: K x Ncols row-major  ->  Wt: Ncols x K row-major (bf16)
__global__ void transpose_to_bf16(const float* __restrict__ W, __bf16* __restrict__ Wt,
                                  int K, int Ncols) {
  size_t t = (size_t)blockIdx.x * blockDim.x + threadIdx.x;
  if (t >= (size_t)K * Ncols) return;
  int k = (int)(t / Ncols), n = (int)(t % Ncols);
  Wt[(size_t)n * K + k] = f2bf(W[t]);
}

// ---------------- WMMA bf16 GEMM:  C[M x Nc] = A[M x K] * Bt[Nc x K]^T ----------------
// one wave per 16x16 tile; 4 waves (64 cols) per block. M % 16 == 0, Nc % 64-or-16 handled by guard.
__global__ void gemm_bf16_wmma(const __bf16* __restrict__ A, const __bf16* __restrict__ Bt,
                               float* __restrict__ C, int M, int K, int Nc) {
  const int lane  = threadIdx.x & 31;
  const int wave  = threadIdx.x >> 5;
  const int mtile = blockIdx.y * 16;
  const int ntile = (blockIdx.x * 4 + wave) * 16;
  if (mtile >= M || ntile >= Nc) return;

  const int m   = mtile + (lane & 15);
  const int n   = ntile + (lane & 15);
  const int k0a = (lane >> 4) * 8;    // A half-wave K offset {0,8}, hi half at +16
  const int k0b = (lane >> 4) * 8;    // B half-wave K offset {0,8}, hi half at +16

  v8f acc = {};
  for (int kt = 0; kt < K; kt += 32) {
    union { v16bf v; uint4 q[2]; } a, b;
    const __bf16* ap = A  + (size_t)m * K + kt + k0a;
    const __bf16* bp = Bt + (size_t)n * K + kt + k0b;
    __builtin_prefetch(ap + 32, 0, 1);      // stream next A chunk (global_prefetch_b8)
    a.q[0] = *(const uint4*)(ap);           // K = k0a .. k0a+7
    a.q[1] = *(const uint4*)(ap + 16);      // K = k0a+16 .. k0a+23
    b.q[0] = *(const uint4*)(bp);           // K = k0b .. k0b+7
    b.q[1] = *(const uint4*)(bp + 16);      // K = k0b+16 .. k0b+23
    acc = __builtin_amdgcn_wmma_f32_16x16x32_bf16(
        /*neg_a=*/false, a.v, /*neg_b=*/false, b.v,
        /*c_mod=*/(short)0, acc, /*reuse_a=*/false, /*reuse_b=*/false);
  }
  // D layout: VGPR r -> M = r (+8 for lanes 16-31), N = lane&15
  const int row0 = mtile + ((lane >> 4) << 3);
  const int col  = ntile + (lane & 15);
#pragma unroll
  for (int r = 0; r < 8; ++r)
    C[(size_t)(row0 + r) * Nc + col] = acc[r];
}

// ---------------- GAT pieces ----------------
// a_src[i,h] = dot(h[i,h,:], att_src[h,:]);  same for a_dst
__global__ void attn_pre(const float* __restrict__ h, const float* __restrict__ att_s,
                         const float* __restrict__ att_d, float* __restrict__ a_s,
                         float* __restrict__ a_d, int n_nodes, int H, int C) {
  size_t t = (size_t)blockIdx.x * blockDim.x + threadIdx.x;
  if (t >= (size_t)n_nodes * H) return;
  int i = (int)(t / H), hh = (int)(t % H);
  const float* row = h + ((size_t)i * H + hh) * C;
  float ss = 0.f, sd = 0.f;
  for (int c = 0; c < C; ++c) { float v = row[c]; ss += v * att_s[hh * C + c]; sd += v * att_d[hh * C + c]; }
  a_s[t] = ss; a_d[t] = sd;
}

__global__ void edge_attn_max(const int* __restrict__ ei, const float* __restrict__ a_s,
                              const float* __restrict__ a_d, unsigned* __restrict__ amaxk,
                              int n_edges, int e_tot, int H) {
  size_t t = (size_t)blockIdx.x * blockDim.x + threadIdx.x;
  if (t >= (size_t)e_tot * H) return;
  int e = (int)(t / H), hh = (int)(t % H);
  int s, d;
  if (e < n_edges) { s = ei[e]; d = ei[n_edges + e]; } else { s = d = e - n_edges; }
  float v = a_s[(size_t)s * H + hh] + a_d[(size_t)d * H + hh];
  v = v > 0.f ? v : NEG_SLOPE * v;                         // leaky_relu
  atomicMax(&amaxk[(size_t)d * H + hh], fkey(v));
}

__global__ void edge_attn_exp(const int* __restrict__ ei, const float* __restrict__ a_s,
                              const float* __restrict__ a_d, const unsigned* __restrict__ amaxk,
                              float* __restrict__ den, float* __restrict__ alpha,
                              int n_edges, int e_tot, int H) {
  size_t t = (size_t)blockIdx.x * blockDim.x + threadIdx.x;
  if (t >= (size_t)e_tot * H) return;
  int e = (int)(t / H), hh = (int)(t % H);
  int s, d;
  if (e < n_edges) { s = ei[e]; d = ei[n_edges + e]; } else { s = d = e - n_edges; }
  float v = a_s[(size_t)s * H + hh] + a_d[(size_t)d * H + hh];
  v = v > 0.f ? v : NEG_SLOPE * v;
  float ex = __expf(v - fdec(amaxk[(size_t)d * H + hh]));
  alpha[t] = ex;
  atomicAdd(&den[(size_t)d * H + hh], ex);
}

// out[dst,h,:] += h[src,h,:] * alpha/(den+eps). 64 consecutive lanes = 64 consecutive channels.
__global__ void edge_aggregate(const int* __restrict__ ei, const float* __restrict__ h,
                               const float* __restrict__ alpha, const float* __restrict__ den,
                               float* __restrict__ out, int n_edges, int e_tot, int H, int C) {
  const int epb = blockDim.x / C;                 // edges per block (4 for C=64)
  const int le  = threadIdx.x / C;
  const int c   = threadIdx.x % C;
  const int e   = blockIdx.x * epb + le;
  if (e >= e_tot) return;
  int s, d;
  if (e < n_edges) { s = ei[e]; d = ei[n_edges + e]; } else { s = d = e - n_edges; }
  for (int hh = 0; hh < H; ++hh) {
    float sc = alpha[(size_t)e * H + hh] / (den[(size_t)d * H + hh] + 1e-16f);
    atomicAdd(&out[((size_t)d * H + hh) * C + c], h[((size_t)s * H + hh) * C + c] * sc);
  }
}

// v = in + bias;  ELU;  -> bf16
__global__ void bias_elu_bf16(const float* __restrict__ in, const float* __restrict__ bias,
                              __bf16* __restrict__ out, size_t n, int F) {
  size_t t = (size_t)blockIdx.x * blockDim.x + threadIdx.x;
  if (t >= n) return;
  float v = in[t] + bias[t % F];
  v = v > 0.f ? v : expm1f(v);
  out[t] = f2bf(v);
}

// global mean pool accumulation
__global__ void pool_kernel(const float* __restrict__ h, const int* __restrict__ batch,
                            float* __restrict__ sums, float* __restrict__ cnt, int n_nodes, int C) {
  size_t t = (size_t)blockIdx.x * blockDim.x + threadIdx.x;
  if (t >= (size_t)n_nodes * C) return;
  int i = (int)(t / C), c = (int)(t % C);
  int g = batch[i];
  atomicAdd(&sums[(size_t)g * C + c], h[t]);
  if (c == 0) atomicAdd(&cnt[g], 1.0f);
}

// graph_emb = sums/cnt + b2; z = relu(emb@W3 + b3); logits = z@W4 + b4
__global__ void head_kernel(const float* __restrict__ sums, const float* __restrict__ cnt,
                            const float* __restrict__ b2, const float* __restrict__ W3,
                            const float* __restrict__ b3, const float* __restrict__ W4,
                            const float* __restrict__ b4, float* __restrict__ out) {
  int g = threadIdx.x;
  if (g >= G_GRAPHS) return;
  float inv = 1.0f / fmaxf(cnt[g], 1.0f);
  float emb[HID];
#pragma unroll
  for (int c = 0; c < HID; ++c) emb[c] = sums[g * HID + c] * inv + b2[c];
  float z[HID / 2];
  for (int j = 0; j < HID / 2; ++j) {
    float acc = b3[j];
    for (int c = 0; c < HID; ++c) acc += emb[c] * W3[c * (HID / 2) + j];
    z[j] = fmaxf(acc, 0.f);
  }
  for (int k = 0; k < 2; ++k) {
    float acc = b4[k];
    for (int j = 0; j < HID / 2; ++j) acc += z[j] * W4[j * 2 + k];
    out[g * 2 + k] = acc;
  }
}

// ---------------- launcher ----------------
extern "C" void kernel_launch(void* const* d_in, const int* in_sizes, int n_in,
                              void* d_out, int out_size, void* d_ws, size_t ws_size,
                              hipStream_t stream) {
  (void)in_sizes; (void)n_in; (void)out_size; (void)ws_size;
  const float* x        = (const float*)d_in[0];
  const int*   ei       = (const int*)  d_in[1];
  const int*   batch    = (const int*)  d_in[2];
  const float* W1       = (const float*)d_in[3];
  const float* att_src1 = (const float*)d_in[4];
  const float* att_dst1 = (const float*)d_in[5];
  const float* b1       = (const float*)d_in[6];
  const float* W2       = (const float*)d_in[7];
  const float* att_src2 = (const float*)d_in[8];
  const float* att_dst2 = (const float*)d_in[9];
  const float* b2       = (const float*)d_in[10];
  const float* W3       = (const float*)d_in[11];
  const float* b3       = (const float*)d_in[12];
  const float* W4       = (const float*)d_in[13];
  const float* b4       = (const float*)d_in[14];
  float* out = (float*)d_out;

  char* ws = (char*)d_ws;
  size_t o = 0;
  auto take = [&](size_t bytes) { size_t r = o; o += (bytes + 255) & ~(size_t)255; return r; };

  __bf16*   xb    = (__bf16*)  (ws + take((size_t)N_NODES * F1 * 2)); // reused as h1b after GEMM1
  __bf16*   W1t   = (__bf16*)  (ws + take((size_t)IN_DIM * F1 * 2));
  __bf16*   W2t   = (__bf16*)  (ws + take((size_t)F1 * HID * 2));
  float*    h1    = (float*)   (ws + take((size_t)N_NODES * F1 * 4)); // reused as h2
  float*    out1  = (float*)   (ws + take((size_t)N_NODES * F1 * 4)); // reused as out2
  float*    a_s   = (float*)   (ws + take((size_t)N_NODES * HEADS * 4));
  float*    a_d   = (float*)   (ws + take((size_t)N_NODES * HEADS * 4));
  unsigned* amaxk = (unsigned*)(ws + take((size_t)N_NODES * HEADS * 4));
  float*    den   = (float*)   (ws + take((size_t)N_NODES * HEADS * 4));
  float*    alpha = (float*)   (ws + take((size_t)E_TOT * HEADS * 4));
  float*    pool  = (float*)   (ws + take((size_t)G_GRAPHS * HID * 4));
  float*    cnt   = (float*)   (ws + take((size_t)G_GRAPHS * 4));

  auto nb = [](size_t n, int bs) { return (unsigned)((n + (size_t)bs - 1) / bs); };

  // ---- prep: precision conversion + accumulator init ----
  cvt_f32_bf16<<<nb((size_t)N_NODES * F1, 256), 256, 0, stream>>>(x, xb, (size_t)N_NODES * F1);
  transpose_to_bf16<<<nb((size_t)IN_DIM * F1, 256), 256, 0, stream>>>(W1, W1t, IN_DIM, F1);
  transpose_to_bf16<<<nb((size_t)F1 * HID, 256), 256, 0, stream>>>(W2, W2t, F1, HID);
  fill_f32<<<nb((size_t)N_NODES * F1, 256), 256, 0, stream>>>(out1, 0.f, (size_t)N_NODES * F1);
  fill_u32<<<nb((size_t)N_NODES * HEADS, 256), 256, 0, stream>>>(amaxk, 0u, (size_t)N_NODES * HEADS);
  fill_f32<<<nb((size_t)N_NODES * HEADS, 256), 256, 0, stream>>>(den, 0.f, (size_t)N_NODES * HEADS);

  // ---- layer 1: h1 = x @ W1 (WMMA bf16, f32 acc) ----
  gemm_bf16_wmma<<<dim3(F1 / 64, N_NODES / 16), 128, 0, stream>>>(xb, W1t, h1, N_NODES, IN_DIM, F1);
  attn_pre<<<nb((size_t)N_NODES * HEADS, 256), 256, 0, stream>>>(h1, att_src1, att_dst1, a_s, a_d, N_NODES, HEADS, HID);
  edge_attn_max<<<nb((size_t)E_TOT * HEADS, 256), 256, 0, stream>>>(ei, a_s, a_d, amaxk, N_EDGES, E_TOT, HEADS);
  edge_attn_exp<<<nb((size_t)E_TOT * HEADS, 256), 256, 0, stream>>>(ei, a_s, a_d, amaxk, den, alpha, N_EDGES, E_TOT, HEADS);
  edge_aggregate<<<(E_TOT + 3) / 4, 256, 0, stream>>>(ei, h1, alpha, den, out1, N_EDGES, E_TOT, HEADS, HID);
  bias_elu_bf16<<<nb((size_t)N_NODES * F1, 256), 256, 0, stream>>>(out1, b1, xb, (size_t)N_NODES * F1, F1);

  // ---- layer 2 (re-init reused accumulators) ----
  fill_f32<<<nb((size_t)N_NODES * HID, 256), 256, 0, stream>>>(out1, 0.f, (size_t)N_NODES * HID);
  fill_u32<<<nb((size_t)N_NODES, 256), 256, 0, stream>>>(amaxk, 0u, (size_t)N_NODES);
  fill_f32<<<nb((size_t)N_NODES, 256), 256, 0, stream>>>(den, 0.f, (size_t)N_NODES);
  gemm_bf16_wmma<<<dim3(HID / 64, N_NODES / 16), 128, 0, stream>>>(xb, W2t, h1, N_NODES, F1, HID);
  attn_pre<<<nb((size_t)N_NODES, 256), 256, 0, stream>>>(h1, att_src2, att_dst2, a_s, a_d, N_NODES, 1, HID);
  edge_attn_max<<<nb((size_t)E_TOT, 256), 256, 0, stream>>>(ei, a_s, a_d, amaxk, N_EDGES, E_TOT, 1);
  edge_attn_exp<<<nb((size_t)E_TOT, 256), 256, 0, stream>>>(ei, a_s, a_d, amaxk, den, alpha, N_EDGES, E_TOT, 1);
  edge_aggregate<<<(E_TOT + 3) / 4, 256, 0, stream>>>(ei, h1, alpha, den, out1, N_EDGES, E_TOT, 1, HID);

  // ---- global mean pool + MLP head ----
  fill_f32<<<nb((size_t)G_GRAPHS * HID, 256), 256, 0, stream>>>(pool, 0.f, (size_t)G_GRAPHS * HID);
  fill_f32<<<1, 64, 0, stream>>>(cnt, 0.f, (size_t)G_GRAPHS);
  pool_kernel<<<nb((size_t)N_NODES * HID, 256), 256, 0, stream>>>(out1, batch, pool, cnt, N_NODES, HID);
  head_kernel<<<1, 64, 0, stream>>>(pool, cnt, b2, W3, b3, W4, b4, out);
}
